// MNT_66795331387551
// MI455X (gfx1250) — compile-verified, matrix operand order checked by reading, not run on hardware
//
#include <hip/hip_runtime.h>
#include <hip/hip_bf16.h>

// ---------------------------------------------------------------------------
// ViT block on gfx1250 (MI455X): all GEMMs via V_WMMA_F32_16X16X32_F16,
// f16 activations/weights, f32 accumulation, fused epilogues.
// Weight (B) panels staged through LDS with GLOBAL_LOAD_ASYNC_TO_LDS_B128
// (ASYNCcnt / s_wait_asynccnt), double-buffered to overlap fetch with WMMA.
// ---------------------------------------------------------------------------

typedef __attribute__((ext_vector_type(16))) _Float16 v16h;
typedef __attribute__((ext_vector_type(8)))  _Float16 v8h;
typedef __attribute__((ext_vector_type(8)))  float    v8f;

#define BATCH     32
#define NTOK      1024            // patches per image (32x32)
#define BN        (BATCH * NTOK)  // 32768 rows
#define DMODEL    256
#define HFF       1024
#define KEMB      768             // 3*16*16
#define NHEADS    8
#define HDIM      32
#define WIN       64
#define NWIN      (NTOK / WIN)    // 16

// --- WMMA tile loader -------------------------------------------------------
// 16-bit A-operand layout (ISA 7.12.2): lane L holds row M=L%16; halves
// j=0..7 -> K = k0 + g*8 + j, j=8..15 -> K = k0 + 16 + g*8 + (j-8), g=L/16.
// B operand (W stored N x K, lane = N) uses the transpose-symmetric layout,
// so the same loader applies.
__device__ inline v16h ldtile(const _Float16* __restrict__ p, int stride,
                              int row, int k0, int g) {
  const _Float16* q = p + (size_t)row * stride + k0 + g * 8;
  v8h lo = *reinterpret_cast<const v8h*>(q);
  v8h hi = *reinterpret_cast<const v8h*>(q + 16);
  v16h r;
#pragma unroll
  for (int i = 0; i < 8; ++i) { r[i] = lo[i]; r[i + 8] = hi[i]; }
  return r;
}

__device__ inline float gelu_tanh(float v) {
  float c = v + 0.044715f * v * v * v;
  return 0.5f * v * (1.f + tanhf(0.7978845608028654f * c));
}

// async global -> LDS b128 (one 16-byte chunk per lane), ASYNCcnt-tracked
__device__ inline void async_ld_b128(unsigned lds_off, const _Float16* gp) {
  asm volatile("global_load_async_to_lds_b128 %0, %1, off"
               :: "v"(lds_off), "v"(gp) : "memory");
}
__device__ inline void wait_async(int n) {
  if (n == 0) asm volatile("s_wait_asynccnt 0x0" ::: "memory");
  else        asm volatile("s_wait_asynccnt 0x1" ::: "memory");
}

// --- generic WMMA GEMM ------------------------------------------------------
// C[M x N] = act(A[M x K] * W[N x K]^T + bias[N]) (+ res[M x N])
// block = 256 threads (8 waves); wave w -> rows [bx*128 + w*16, +16),
// cols [by*64, +64) as 4 accumulator tiles. B panel (64x32 f16 = 4KB) is
// async-staged into LDS once per block per k-step, double-buffered.
template <int ACT, bool HAS_RES, bool OUT_F32, bool OUT_F16>
__global__ __launch_bounds__(256) void gemm_wmma(
    const _Float16* __restrict__ A, const _Float16* __restrict__ W,
    const float* __restrict__ bias, const float* __restrict__ res,
    float* __restrict__ outf, _Float16* __restrict__ outh,
    int M, int N, int K) {
  __shared__ __align__(16) _Float16 Bst[2][64 * 32];  // double-buffered panel

  const int lane = threadIdx.x & 31;
  const int wave = threadIdx.x >> 5;
  const int g    = lane >> 4;
  const int l15  = lane & 15;
  const int m0   = blockIdx.x * 128 + wave * 16;
  const int n0   = blockIdx.y * 64;

  // this thread's slice of the B panel: 16 bytes of row (n0+n), k-slice kk
  const int bt_n  = threadIdx.x >> 2;         // 0..63
  const int bt_k  = (threadIdx.x & 3) * 8;    // 0,8,16,24
  const _Float16* wrow = W + (size_t)(n0 + bt_n) * K + bt_k;

  v8f acc[4] = {v8f{}, v8f{}, v8f{}, v8f{}};

  const int nsteps = K / 32;
  {
    unsigned lds0 = (unsigned)(uintptr_t)&Bst[0][bt_n * 32 + bt_k];
    async_ld_b128(lds0, wrow);                 // prologue: panel for k0=0
  }
  for (int s = 0; s < nsteps; ++s) {
    const int k0 = s * 32;
    const bool more = (s + 1 < nsteps);
    if (more) {
      unsigned ldsn =
          (unsigned)(uintptr_t)&Bst[(s + 1) & 1][bt_n * 32 + bt_k];
      async_ld_b128(ldsn, wrow + k0 + 32);     // prefetch next panel
    }
    wait_async(more ? 1 : 0);                  // current panel landed
    __syncthreads();

    v16h a = ldtile(A, K, m0 + l15, k0, g);
    const _Float16* bp = &Bst[s & 1][0];
#pragma unroll
    for (int t = 0; t < 4; ++t) {
      v16h b = ldtile(bp, 32, t * 16 + l15, 0, g);
      acc[t] = __builtin_amdgcn_wmma_f32_16x16x32_f16(
          false, a, false, b, (short)0, acc[t], false, false);
    }
    __syncthreads();                           // reads done before overwrite
  }

#pragma unroll
  for (int t = 0; t < 4; ++t) {
    const int col = n0 + t * 16 + l15;
    const float bv = bias[col];
#pragma unroll
    for (int r = 0; r < 8; ++r) {
      const int row = m0 + r + 8 * g;
      float v = acc[t][r] + bv;
      if (ACT == 1) v = gelu_tanh(v);
      const size_t o = (size_t)row * N + col;
      if (HAS_RES) v += res[o];
      if (OUT_F32) outf[o] = v;
      if (OUT_F16) outh[o] = (_Float16)v;
    }
  }
}

// --- weight / input prep ----------------------------------------------------
__global__ __launch_bounds__(256) void cvt_f32_f16(
    const float* __restrict__ in, _Float16* __restrict__ out, int n) {
  int i = blockIdx.x * 256 + threadIdx.x;
  if (i < n) out[i] = (_Float16)in[i];
}

// patches: x(B,3,512,512) -> A(BN, 768), feature = c*256 + py*16 + px
__global__ __launch_bounds__(256) void patchify(
    const float* __restrict__ x, _Float16* __restrict__ A) {
  int idx = blockIdx.x * 256 + threadIdx.x;
  if (idx >= BN * KEMB) return;
  int f = idx % KEMB, bn = idx / KEMB;
  int c = f >> 8, rem = f & 255, py = rem >> 4, px = rem & 15;
  int b = bn >> 10, n = bn & 1023, nh = n >> 5, nw = n & 31;
  A[idx] = (_Float16)x[(((size_t)(b * 3 + c) * 512) + nh * 16 + py) * 512 +
                       nw * 16 + px];
}

// --- LayerNorm + sinusoidal PE (in-place f32, plus f16 copy) ----------------
__global__ __launch_bounds__(256) void ln_pe(
    float* __restrict__ h, _Float16* __restrict__ h16,
    const float* __restrict__ lnw, const float* __restrict__ lnb) {
  __shared__ float red[256];
  const int row = blockIdx.x, d = threadIdx.x;
  const size_t o = (size_t)row * DMODEL + d;
  float x = h[o];
  red[d] = x; __syncthreads();
  for (int s = 128; s > 0; s >>= 1) {
    if (d < s) red[d] += red[d + s];
    __syncthreads();
  }
  float mu = red[0] * (1.f / DMODEL);
  __syncthreads();
  float xc = x - mu;
  red[d] = xc * xc; __syncthreads();
  for (int s = 128; s > 0; s >>= 1) {
    if (d < s) red[d] += red[d + s];
    __syncthreads();
  }
  float var = red[0] * (1.f / DMODEL);
  float y = xc * rsqrtf(var + 1e-5f) * lnw[d] + lnb[d];
  // sinusoidal PE: i = d&~1, div = exp(-ln(1e4)*i/256)
  int n = row & (NTOK - 1);
  float div = __expf(-0.03597789207859278f * (float)(d & ~1));
  float ang = (float)n * div;
  y += (d & 1) ? cosf(ang) : sinf(ang);
  h[o] = y;
  h16[o] = (_Float16)y;
}

// --- windowed attention -----------------------------------------------------
__device__ inline float rmax16(float v) {
  v = fmaxf(v, __shfl_xor(v, 1, 32));
  v = fmaxf(v, __shfl_xor(v, 2, 32));
  v = fmaxf(v, __shfl_xor(v, 4, 32));
  v = fmaxf(v, __shfl_xor(v, 8, 32));
  return v;
}
__device__ inline float rsum16(float v) {
  v += __shfl_xor(v, 1, 32);
  v += __shfl_xor(v, 2, 32);
  v += __shfl_xor(v, 4, 32);
  v += __shfl_xor(v, 8, 32);
  return v;
}

// grid (NWIN, NHEADS, BATCH), block 128 (4 waves, one 16-query strip each)
__global__ __launch_bounds__(128) void attn_win(
    const _Float16* __restrict__ qkv, _Float16* __restrict__ out) {
  __shared__ __align__(16) _Float16 Vt[HDIM][80];      // V transposed [d][key]
  __shared__ __align__(16) _Float16 P[4][16][80];      // softmax probs / wave

  const int lane = threadIdx.x & 31;
  const int qi   = threadIdx.x >> 5;   // wave id = query tile
  const int g    = lane >> 4;
  const int l15  = lane & 15;
  const int win  = blockIdx.x, head = blockIdx.y, b = blockIdx.z;
  const int bn0  = b * NTOK + win * WIN;

  // stage V transposed into LDS: thread t loads 16 dims of one key
  {
    const int t = threadIdx.x;
    const int key = t & 63, dg = t >> 6;
    const _Float16* vp =
        qkv + (size_t)(bn0 + key) * (3 * DMODEL) + 2 * DMODEL + head * HDIM +
        dg * 16;
    v8h lo = *reinterpret_cast<const v8h*>(vp);
    v8h hi = *reinterpret_cast<const v8h*>(vp + 8);
#pragma unroll
    for (int j = 0; j < 8; ++j) {
      Vt[dg * 16 + j][key] = lo[j];
      Vt[dg * 16 + 8 + j][key] = hi[j];
    }
  }
  __syncthreads();

  // S = Q K^T  (K-dim = 32 -> single WMMA per tile pair)
  const _Float16* qbase = qkv + head * HDIM;
  const _Float16* kbase = qkv + DMODEL + head * HDIM;
  v16h qa = ldtile(qbase, 3 * DMODEL, bn0 + qi * 16 + l15, 0, g);
  v8f s[4];
#pragma unroll
  for (int kj = 0; kj < 4; ++kj) {
    v16h kb = ldtile(kbase, 3 * DMODEL, bn0 + kj * 16 + l15, 0, g);
    v8f z = {};
    s[kj] = __builtin_amdgcn_wmma_f32_16x16x32_f16(
        false, qa, false, kb, (short)0, z, false, false);
  }

  // softmax over 64 keys per query row; row r+8g lives in one 16-lane group
  const float scale = 0.17677669529663687f;  // 1/sqrt(32)
#pragma unroll
  for (int r = 0; r < 8; ++r) {
    float v0 = s[0][r] * scale, v1 = s[1][r] * scale;
    float v2 = s[2][r] * scale, v3 = s[3][r] * scale;
    float m = rmax16(fmaxf(fmaxf(v0, v1), fmaxf(v2, v3)));
    float p0 = __expf(v0 - m), p1 = __expf(v1 - m);
    float p2 = __expf(v2 - m), p3 = __expf(v3 - m);
    float inv = 1.f / rsum16(p0 + p1 + p2 + p3);
    const int prow = r + 8 * g;
    P[qi][prow][0 * 16 + l15] = (_Float16)(p0 * inv);
    P[qi][prow][1 * 16 + l15] = (_Float16)(p1 * inv);
    P[qi][prow][2 * 16 + l15] = (_Float16)(p2 * inv);
    P[qi][prow][3 * 16 + l15] = (_Float16)(p3 * inv);
  }
  __syncthreads();

  // O = P V   (K-dim = 64 -> 2 steps; N = 32 -> 2 tiles)
  v8f o[2] = {v8f{}, v8f{}};
#pragma unroll
  for (int ks = 0; ks < 2; ++ks) {
    v16h pa = ldtile(&P[qi][0][0], 80, l15, ks * 32, g);
#pragma unroll
    for (int nj = 0; nj < 2; ++nj) {
      v16h vb = ldtile(&Vt[0][0], 80, nj * 16 + l15, ks * 32, g);
      o[nj] = __builtin_amdgcn_wmma_f32_16x16x32_f16(
          false, pa, false, vb, (short)0, o[nj], false, false);
    }
  }

#pragma unroll
  for (int nj = 0; nj < 2; ++nj) {
#pragma unroll
    for (int r = 0; r < 8; ++r) {
      const int token = bn0 + qi * 16 + r + 8 * g;
      const int col = head * HDIM + nj * 16 + l15;
      out[(size_t)token * DMODEL + col] = (_Float16)o[nj][r];
    }
  }
}

// ---------------------------------------------------------------------------
extern "C" void kernel_launch(void* const* d_in, const int* in_sizes, int n_in,
                              void* d_out, int out_size, void* d_ws,
                              size_t ws_size, hipStream_t stream) {
  const float* x      = (const float*)d_in[0];
  const float* conv_w = (const float*)d_in[1];
  const float* conv_b = (const float*)d_in[2];
  const float* ln_w   = (const float*)d_in[3];
  const float* ln_b   = (const float*)d_in[4];
  const float* qkv_w  = (const float*)d_in[5];
  const float* qkv_b  = (const float*)d_in[6];
  const float* proj_w = (const float*)d_in[7];
  const float* proj_b = (const float*)d_in[8];
  const float* ff_w1  = (const float*)d_in[9];
  const float* ff_b1  = (const float*)d_in[10];
  const float* ff_w2  = (const float*)d_in[11];
  const float* ff_b2  = (const float*)d_in[12];
  float* out = (float*)d_out;

  char* ws = (char*)d_ws;
  size_t off = 0;
  auto alloc = [&](size_t bytes) {
    size_t p = off;
    off += (bytes + 255) & ~(size_t)255;
    return p;
  };
  // f16 weights
  size_t oWE = alloc((size_t)DMODEL * KEMB * 2);
  size_t oW1 = alloc((size_t)HFF * DMODEL * 2);
  size_t oW2 = alloc((size_t)DMODEL * HFF * 2);
  size_t oWQ = alloc((size_t)(3 * DMODEL) * DMODEL * 2);
  size_t oWP = alloc((size_t)DMODEL * DMODEL * 2);
  // activations (buffers reused across stages)
  size_t oA   = alloc((size_t)BN * KEMB * 2);    // patches f16, later qkv f16
  size_t oH0f = alloc((size_t)BN * DMODEL * 4);  // embed f32 -> h0 f32
  size_t oH16 = alloc((size_t)BN * DMODEL * 2);  // h0 f16 -> h3 f16
  size_t oHid = alloc((size_t)BN * HFF * 2);     // FFN hidden f16
  size_t oF16 = alloc((size_t)BN * DMODEL * 2);  // FFN out f16 -> attn out f16
  size_t oH3f = alloc((size_t)BN * DMODEL * 4);  // h3 f32

  _Float16* wE = (_Float16*)(ws + oWE);
  _Float16* w1 = (_Float16*)(ws + oW1);
  _Float16* w2 = (_Float16*)(ws + oW2);
  _Float16* wQ = (_Float16*)(ws + oWQ);
  _Float16* wP = (_Float16*)(ws + oWP);
  _Float16* Abig = (_Float16*)(ws + oA);
  float*    h0f  = (float*)(ws + oH0f);
  _Float16* h16  = (_Float16*)(ws + oH16);
  _Float16* hid  = (_Float16*)(ws + oHid);
  _Float16* f16b = (_Float16*)(ws + oF16);
  float*    h3f  = (float*)(ws + oH3f);

  auto cvt = [&](const float* src, _Float16* dst, int n) {
    cvt_f32_f16<<<(n + 255) / 256, 256, 0, stream>>>(src, dst, n);
  };
  cvt(conv_w, wE, DMODEL * KEMB);
  cvt(ff_w1, w1, HFF * DMODEL);
  cvt(ff_w2, w2, DMODEL * HFF);
  cvt(qkv_w, wQ, 3 * DMODEL * DMODEL);
  cvt(proj_w, wP, DMODEL * DMODEL);

  // patchify x -> A (BN x 768 f16)
  patchify<<<(BN * KEMB + 255) / 256, 256, 0, stream>>>(x, Abig);

  const dim3 blk(256);
  // 1) patch embed: h0f = A * wE^T + conv_b        (N=256, K=768)
  gemm_wmma<0, false, true, false>
      <<<dim3(BN / 128, DMODEL / 64), blk, 0, stream>>>(
          Abig, wE, conv_b, nullptr, h0f, nullptr, BN, DMODEL, KEMB);
  // 2) LayerNorm + PE (in place on h0f, f16 copy in h16)
  ln_pe<<<BN, 256, 0, stream>>>(h0f, h16, ln_w, ln_b);
  // 3) FFN1a: hid = gelu(h16 * w1^T + b1)          (N=1024, K=256)
  gemm_wmma<1, false, false, true>
      <<<dim3(BN / 128, HFF / 64), blk, 0, stream>>>(
          h16, w1, ff_b1, nullptr, nullptr, hid, BN, HFF, DMODEL);
  // 4) FFN1b: f16b = hid * w2^T + b2               (N=256, K=1024)
  gemm_wmma<0, false, false, true>
      <<<dim3(BN / 128, DMODEL / 64), blk, 0, stream>>>(
          hid, w2, ff_b2, nullptr, nullptr, f16b, BN, DMODEL, HFF);
  // 5) QKV: Abig = f16b * wQ^T + qkv_b             (N=768, K=256)
  gemm_wmma<0, false, false, true>
      <<<dim3(BN / 128, (3 * DMODEL) / 64), blk, 0, stream>>>(
          f16b, wQ, qkv_b, nullptr, nullptr, Abig, BN, 3 * DMODEL, DMODEL);
  // 6) windowed attention: f16b = attn(Abig)
  attn_win<<<dim3(NWIN, NHEADS, BATCH), 128, 0, stream>>>(Abig, f16b);
  // 7) proj + residual h0: h3f/h16 = f16b * wP^T + proj_b + h0f
  gemm_wmma<0, true, true, true>
      <<<dim3(BN / 128, DMODEL / 64), blk, 0, stream>>>(
          f16b, wP, proj_b, h0f, h3f, h16, BN, DMODEL, DMODEL);
  // 8) FFN2a: hid = gelu(h16 * w1^T + b1)
  gemm_wmma<1, false, false, true>
      <<<dim3(BN / 128, HFF / 64), blk, 0, stream>>>(
          h16, w1, ff_b1, nullptr, nullptr, hid, BN, HFF, DMODEL);
  // 9) FFN2b + residual h3 -> out (f32)
  gemm_wmma<0, true, true, false>
      <<<dim3(BN / 128, DMODEL / 64), blk, 0, stream>>>(
          hid, w2, ff_b2, h3f, out, nullptr, BN, DMODEL, HFF);
}
